// Critic_46514495816201
// MI455X (gfx1250) — matrix operand order repaired and checked by reading, not compiled, not run on hardware
//
#include <hip/hip_runtime.h>
#include <hip/hip_bf16.h>
#include <math.h>

typedef __attribute__((ext_vector_type(16))) _Float16 v16h;
typedef __attribute__((ext_vector_type(8)))  _Float16 v8h;
typedef __attribute__((ext_vector_type(8)))  float    v8f;

#define N_NODES 100000
#define N_EDGES 3200000
#define HID 64
// GEMM blocks cover 128 rows; pad node-major scratch to a multiple of 128 so
// the WMMA kernel needs no bounds checks at all (tail rows are scratch-only
// and never read by downstream kernels).
#define N_PAD 100096

// ---------------------------------------------------------------------------
// Stage 0: degree / normalization
// ---------------------------------------------------------------------------
__global__ void k_deg_init(float* __restrict__ deg) {
    int i = blockIdx.x * blockDim.x + threadIdx.x;
    if (i < N_NODES) deg[i] = 1.0f;               // self-loop weight
}

__global__ void k_deg_scatter(float* __restrict__ deg,
                              const int* __restrict__ col,
                              const float* __restrict__ w) {
    int e = blockIdx.x * blockDim.x + threadIdx.x;
    if (e < N_EDGES) atomicAdd(&deg[col[e]], w[e]);
}

__global__ void k_dinv(float* __restrict__ deg) {
    int i = blockIdx.x * blockDim.x + threadIdx.x;
    if (i < N_NODES) {
        float d = deg[i];
        deg[i] = (d > 0.0f) ? rsqrtf(d) : 0.0f;   // in-place: becomes dinv
    }
}

__global__ void k_norm(float* __restrict__ norm,
                       const float* __restrict__ dinv,
                       const int* __restrict__ row,
                       const int* __restrict__ col,
                       const float* __restrict__ w) {
    int e = blockIdx.x * blockDim.x + threadIdx.x;
    if (e < N_EDGES) norm[e] = dinv[row[e]] * w[e] * dinv[col[e]];
}

// ---------------------------------------------------------------------------
// Layer-1 GEMM (K = 6, degenerate shape -> VALU)
// ---------------------------------------------------------------------------
__global__ void k_gemm_k6(const float* __restrict__ vf,   // [N,6]
                          const float* __restrict__ W,    // [6,64]
                          float* __restrict__ h) {        // [N,64]
    int t = blockIdx.x * blockDim.x + threadIdx.x;
    if (t >= N_NODES * HID) return;
    int n = t >> 6, c = t & 63;
    const float* x = vf + (size_t)n * 6;
    float acc = 0.0f;
#pragma unroll
    for (int k = 0; k < 6; ++k) acc = fmaf(x[k], W[k * HID + c], acc);
    h[t] = acc;
}

// ---------------------------------------------------------------------------
// WMMA GEMM for layers 2-4: h[N,64] = x16[N,64] @ W[64,64]
//
// Block = 256 threads = 8 waves. W (f32) is converted to f16 and staged
// TRANSPOSED in LDS once per block; each wave owns one 16-col n-tile and
// sweeps 4 m-tiles (64 rows), keeping its two B fragments in registers:
//   - B fragment build: 2x ds_load_b128 per K-block (two 8-half chunks, 16 halfs apart)
//   - A fragment build: 2x global_load_b128 per K-block (one 4-load clause)
//   - 8x v_wmma_f32_16x16x32_f16 per wave, branchless stores
// Buffers are padded to N_PAD rows so no bounds checks are needed.
// ---------------------------------------------------------------------------
__global__ void __launch_bounds__(256)
k_gemm_wmma(const _Float16* __restrict__ A,   // [N_PAD,64] row-major f16
            const float* __restrict__ W,      // [64,64] row-major f32
            float* __restrict__ C) {          // [N_PAD,64]
    __shared__ __align__(16) _Float16 Wt[HID * HID];   // Wt[n][k] = W[k][n], 8KB

    const int t = threadIdx.x;
    // Cooperative f32->f16 transpose: each thread converts 16 contiguous
    // source elements (coalesced float4 reads), scattered b16 LDS writes.
#pragma unroll
    for (int i = 0; i < 16; ++i) {
        int idx = t * 16 + i;                 // 256*16 = 4096 = HID*HID
        int k = idx >> 6, n = idx & 63;
        Wt[n * HID + k] = (_Float16)W[idx];
    }
    __syncthreads();

    const int lane   = t & 31;                // wave32
    const int waveId = t >> 5;                // 0..7
    const int half   = lane >> 4;
    const int l16    = lane & 15;
    const int n0     = (waveId & 3) * 16;                       // n-tile
    const int m0     = (blockIdx.x * 2 + (waveId >> 2)) * 64;   // 64-row strip

    // B fragments (held across the whole M sweep).
    // Fragment elem j <-> K = kb + 8*half + j (j<8), kb + 16 + 8*half + (j-8) (j>=8), col = n0 + l16.
    const _Float16* bp = &Wt[(n0 + l16) * HID + 8 * half];
    v8h b00 = *(const v8h*)(bp + 0);
    v8h b01 = *(const v8h*)(bp + 16);
    v8h b10 = *(const v8h*)(bp + 32);
    v8h b11 = *(const v8h*)(bp + 48);
    v16h bf0 = __builtin_shufflevector(b00, b01, 0,1,2,3,4,5,6,7,8,9,10,11,12,13,14,15);
    v16h bf1 = __builtin_shufflevector(b10, b11, 0,1,2,3,4,5,6,7,8,9,10,11,12,13,14,15);

#pragma unroll
    for (int mt = 0; mt < 4; ++mt) {
        const int m = m0 + mt * 16;
        // A fragments: lane half h holds K = 8h..8h+7 and 16+8h..16+8h+7
        // (two contiguous 16B chunks per K-block). Unconditional: buffer is
        // padded to N_PAD rows.
        const _Float16* ap = A + (size_t)(m + l16) * HID + 8 * half;
        v8h x0 = *(const v8h*)(ap +  0);
        v8h x1 = *(const v8h*)(ap + 16);
        v8h x2 = *(const v8h*)(ap + 32);
        v8h x3 = *(const v8h*)(ap + 48);
        v16h a0 = __builtin_shufflevector(x0, x1, 0,1,2,3,4,5,6,7,8,9,10,11,12,13,14,15);
        v16h a1 = __builtin_shufflevector(x2, x3, 0,1,2,3,4,5,6,7,8,9,10,11,12,13,14,15);

        v8f acc = {};
        acc = __builtin_amdgcn_wmma_f32_16x16x32_f16(false, a0, false, bf0,
                                                     (short)0, acc, false, false);
        acc = __builtin_amdgcn_wmma_f32_16x16x32_f16(false, a1, false, bf1,
                                                     (short)0, acc, false, false);
        // C/D layout: VGPR r -> row M = 8*half + r, col N = lane%16.
        // One base address; 8 stores at 256B immediate offsets.
        float* cp = C + (size_t)(m + 8 * half) * HID + n0 + l16;
#pragma unroll
        for (int r = 0; r < 8; ++r) {
            cp[(size_t)r * HID] = acc[r];
        }
    }
}

// ---------------------------------------------------------------------------
// Aggregation: self-loop init + edge scatter-add (L2-resident atomics)
// ---------------------------------------------------------------------------
__global__ void k_agg_init(float* __restrict__ agg,
                           const float* __restrict__ h,
                           const float* __restrict__ dinv) {
    int t = blockIdx.x * blockDim.x + threadIdx.x;
    if (t >= N_NODES * HID) return;
    float di = dinv[t >> 6];
    agg[t] = h[t] * di * di;                  // self-loop: norm = dinv^2
}

__global__ void k_edge_scatter(float* __restrict__ agg,
                               const float* __restrict__ h,
                               const float* __restrict__ norm,
                               const int* __restrict__ row,
                               const int* __restrict__ col) {
    long long t = (long long)blockIdx.x * blockDim.x + threadIdx.x;
    if (t >= (long long)N_EDGES * 16) return;
    int e  = (int)(t >> 4);
    int c4 = (int)(t & 15) * 4;
    float nv = norm[e];
    const float4 v = *reinterpret_cast<const float4*>(h + (size_t)row[e] * HID + c4);
    float* dst = agg + (size_t)col[e] * HID + c4;
    atomicAdd(dst + 0, v.x * nv);
    atomicAdd(dst + 1, v.y * nv);
    atomicAdd(dst + 2, v.z * nv);
    atomicAdd(dst + 3, v.w * nv);
}

__global__ void k_bias_sigmoid(float* __restrict__ x,
                               _Float16* __restrict__ x16,
                               const float* __restrict__ agg,
                               const float* __restrict__ bias) {
    int t = blockIdx.x * blockDim.x + threadIdx.x;
    if (t >= N_NODES * HID) return;
    float z = agg[t] + bias[t & 63];
    float v = 1.0f / (1.0f + __expf(-z));
    x[t]   = v;
    x16[t] = (_Float16)v;                     // next layer's WMMA A operand
}

// ---------------------------------------------------------------------------
// Layer 5 (dout = 1) + global mean pool
// ---------------------------------------------------------------------------
__global__ void k_gemm_out(const float* __restrict__ x,   // [N,64]
                           const float* __restrict__ W5,  // [64]
                           float* __restrict__ h5) {      // [N]
    int n = blockIdx.x * blockDim.x + threadIdx.x;
    if (n >= N_NODES) return;
    const float4* xr = reinterpret_cast<const float4*>(x + (size_t)n * HID);
    const float4* wr = reinterpret_cast<const float4*>(W5);
    float acc = 0.0f;
#pragma unroll
    for (int k = 0; k < 16; ++k) {
        float4 a = xr[k], b = wr[k];
        acc = fmaf(a.x, b.x, acc);
        acc = fmaf(a.y, b.y, acc);
        acc = fmaf(a.z, b.z, acc);
        acc = fmaf(a.w, b.w, acc);
    }
    h5[n] = acc;
}

__global__ void k_agg_init_1(float* __restrict__ agg5,
                             const float* __restrict__ h5,
                             const float* __restrict__ dinv) {
    int n = blockIdx.x * blockDim.x + threadIdx.x;
    if (n < N_NODES) {
        float di = dinv[n];
        agg5[n] = h5[n] * di * di;
    }
}

__global__ void k_edge_scatter_1(float* __restrict__ agg5,
                                 const float* __restrict__ h5,
                                 const float* __restrict__ norm,
                                 const int* __restrict__ row,
                                 const int* __restrict__ col) {
    int e = blockIdx.x * blockDim.x + threadIdx.x;
    if (e < N_EDGES) atomicAdd(&agg5[col[e]], h5[row[e]] * norm[e]);
}

__global__ void k_zero_out(float* __restrict__ out) {
    if (threadIdx.x == 0 && blockIdx.x == 0) out[0] = 0.0f;
}

__global__ void k_final_reduce(const float* __restrict__ agg5,
                               const float* __restrict__ b5,
                               float* __restrict__ out) {
    __shared__ float sdata[256];
    int n = blockIdx.x * blockDim.x + threadIdx.x;
    float v = 0.0f;
    if (n < N_NODES) {
        float z = agg5[n] + b5[0];
        v = 1.0f / (1.0f + __expf(-z));
    }
    sdata[threadIdx.x] = v;
    __syncthreads();
    for (int s = 128; s > 0; s >>= 1) {
        if (threadIdx.x < (unsigned)s) sdata[threadIdx.x] += sdata[threadIdx.x + s];
        __syncthreads();
    }
    if (threadIdx.x == 0) atomicAdd(out, sdata[0] * (1.0f / (float)N_NODES));
}

// ---------------------------------------------------------------------------
// Host orchestration
// ---------------------------------------------------------------------------
extern "C" void kernel_launch(void* const* d_in, const int* in_sizes, int n_in,
                              void* d_out, int out_size, void* d_ws, size_t ws_size,
                              hipStream_t stream) {
    const float* vf   = (const float*)d_in[0];           // [N,6]
    const int*   edges= (const int*)d_in[1];             // [2,E]
    const float* wts  = (const float*)d_in[2];           // [E]
    const float* W1 = (const float*)d_in[3];  const float* b1 = (const float*)d_in[4];
    const float* W2 = (const float*)d_in[5];  const float* b2 = (const float*)d_in[6];
    const float* W3 = (const float*)d_in[7];  const float* b3 = (const float*)d_in[8];
    const float* W4 = (const float*)d_in[9];  const float* b4 = (const float*)d_in[10];
    const float* W5 = (const float*)d_in[11]; const float* b5 = (const float*)d_in[12];
    float* out = (float*)d_out;

    const int* row = edges;            // edges[0,:]
    const int* col = edges + N_EDGES;  // edges[1,:]

    // carve workspace (256B-aligned slices)
    char* ws = (char*)d_ws;
    size_t off = 0;
    auto carve = [&](size_t bytes) -> char* {
        off = (off + 255) & ~(size_t)255;
        char* p = ws + off;
        off += bytes;
        return p;
    };
    float*    dinv  = (float*)   carve((size_t)N_NODES * 4);
    float*    norm  = (float*)   carve((size_t)N_EDGES * 4);
    float*    xbuf  = (float*)   carve((size_t)N_NODES * HID * 4);
    float*    hbuf  = (float*)   carve((size_t)N_PAD  * HID * 4);  // padded (WMMA C)
    float*    aggb  = (float*)   carve((size_t)N_NODES * HID * 4);
    _Float16* x16   = (_Float16*)carve((size_t)N_PAD  * HID * 2);  // padded (WMMA A)
    float*    h5    = (float*)   carve((size_t)N_NODES * 4);
    float*    agg5  = (float*)   carve((size_t)N_NODES * 4);
    (void)ws_size; (void)in_sizes; (void)n_in; (void)out_size;

    const int B = 256;
    const int gN   = (N_NODES + B - 1) / B;
    const int gE   = (N_EDGES + B - 1) / B;
    const int gNH  = (N_NODES * HID + B - 1) / B;
    const long long e16 = (long long)N_EDGES * 16;
    const int gE16 = (int)((e16 + B - 1) / B);
    const int gGemm = N_PAD / 128;             // 128 rows per block (2 strips)

    // gcn_norm
    k_deg_init   <<<gN, B, 0, stream>>>(dinv);
    k_deg_scatter<<<gE, B, 0, stream>>>(dinv, col, wts);
    k_dinv       <<<gN, B, 0, stream>>>(dinv);
    k_norm       <<<gE, B, 0, stream>>>(norm, dinv, row, col, wts);

    // ----- layer 1 (K=6 VALU GEMM) -----
    k_gemm_k6     <<<gNH, B, 0, stream>>>(vf, W1, hbuf);
    k_agg_init    <<<gNH, B, 0, stream>>>(aggb, hbuf, dinv);
    k_edge_scatter<<<gE16, B, 0, stream>>>(aggb, hbuf, norm, row, col);
    k_bias_sigmoid<<<gNH, B, 0, stream>>>(xbuf, x16, aggb, b1);

    // ----- layers 2-4 (WMMA GEMM) -----
    const float* Ws[3] = {W2, W3, W4};
    const float* bs[3] = {b2, b3, b4};
    for (int l = 0; l < 3; ++l) {
        k_gemm_wmma   <<<gGemm, B, 0, stream>>>(x16, Ws[l], hbuf);
        k_agg_init    <<<gNH, B, 0, stream>>>(aggb, hbuf, dinv);
        k_edge_scatter<<<gE16, B, 0, stream>>>(aggb, hbuf, norm, row, col);
        k_bias_sigmoid<<<gNH, B, 0, stream>>>(xbuf, x16, aggb, bs[l]);
    }

    // ----- layer 5 (dout=1) + mean pool -----
    k_gemm_out      <<<gN, B, 0, stream>>>(xbuf, W5, h5);
    k_agg_init_1    <<<gN, B, 0, stream>>>(agg5, h5, dinv);
    k_edge_scatter_1<<<gE, B, 0, stream>>>(agg5, h5, norm, row, col);
    k_zero_out      <<<1, 32, 0, stream>>>(out);
    k_final_reduce  <<<gN, B, 0, stream>>>(agg5, b5, out);
}
